// SupRobConLoss_4672924418296
// MI455X (gfx1250) — matrix operand order repaired and checked by reading, not compiled, not run on hardware
//
#include <hip/hip_runtime.h>
#include <hip/hip_bf16.h>

typedef __attribute__((ext_vector_type(16))) _Float16 v16h;
typedef __attribute__((ext_vector_type(8)))  _Float16 v8h;
typedef __attribute__((ext_vector_type(4)))  _Float16 v4h;
typedef __attribute__((ext_vector_type(8)))  float    v8f;

#define BS    2048
#define DIM   512
#define NTOT  4096
#define INV_T (1.0f / 0.07f)

// Detect the gfx1250 Tensor Data Mover builtins (device pass only).
#if defined(__has_builtin)
#  if __has_builtin(__builtin_amdgcn_tensor_load_to_lds) && __has_builtin(__builtin_amdgcn_s_wait_tensorcnt)
#    define USE_TDM 1
#  endif
#endif
#ifndef USE_TDM
#  define USE_TDM 0
#endif

#if USE_TDM
typedef unsigned int u32x4 __attribute__((ext_vector_type(4)));
typedef int          i32x8 __attribute__((ext_vector_type(8)));
typedef int          i32x4 __attribute__((ext_vector_type(4)));

// TDM: DMA one contiguous 16 KB tile (8192 fp16) from global into LDS.
// D# per ISA cdna5 ch.8: 1-D tensor, data_size=2B, tile_dim0=tensor_dim0=8192.
static __device__ __forceinline__ void tdm_stage_16k(const _Float16* gsrc,
                                                     const _Float16* ldst) {
  u32x4 g0;
  g0[0] = 1u;                                             // count=1 valid user D#
  g0[1] = (unsigned)(unsigned long long)(uintptr_t)ldst;  // lds_addr (bytes)
  unsigned long long ga = (unsigned long long)(uintptr_t)gsrc;
  g0[2] = (unsigned)ga;                                   // global_addr[31:0]
  g0[3] = ((unsigned)(ga >> 32) & 0x01FFFFFFu) | (2u << 30); // addr[56:32] | type=2
  i32x8 g1;
  g1[0] = 1 << 16;                 // workgroup_mask=0, data_size=1 (2 bytes)
  g1[1] = (int)(8192u << 16);      // tensor_dim0[15:0]=8192 into bits[63:48]
  g1[2] = (int)(1u << 16);         // tensor_dim0 hi=0 ; tensor_dim1=1
  g1[3] = (int)(8192u << 16);      // tensor_dim1 hi=0 ; tile_dim0=8192
  g1[4] = 1;                       // tile_dim1=1 ; tile_dim2=0
  g1[5] = 8192;                    // tensor_dim0_stride=8192
  g1[6] = 0;
  g1[7] = 0;
  i32x4 z4 = (i32x4)0;
#if __clang_major__ >= 23
  i32x8 z8 = (i32x8)0;
  __builtin_amdgcn_tensor_load_to_lds(g0, g1, z4, z4, z8, 0);
#else
  __builtin_amdgcn_tensor_load_to_lds(g0, g1, z4, z4, 0);
#endif
}
#endif  // USE_TDM

// Fallback cooperative stage: 32 KB (both matrices) by 256 threads.
static __device__ __forceinline__ void stage_copy(const _Float16* s0,
                                                  const _Float16* s1,
                                                  _Float16* dst, int tid) {
  const float4* a = (const float4*)s0;
  const float4* b = (const float4*)s1;
  float4* d = (float4*)dst;
  #pragma unroll
  for (int k = 0; k < 4; ++k) {
    d[tid + k * 256]        = a[tid + k * 256];
    d[tid + k * 256 + 1024] = b[tid + k * 256];
  }
}

// ---------------------------------------------------------------------------
// Phase 1: row-normalize z0/z1/z1_adv -> fp16 vn01 / vn01_adv in workspace.
// Blocks 0..95 also zero the 12288-float S/P/C partial arrays.
// ---------------------------------------------------------------------------
__global__ __launch_bounds__(128)
void sup_normalize(const float* __restrict__ z0, const float* __restrict__ z1,
                   const float* __restrict__ z1a,
                   _Float16* __restrict__ vn, _Float16* __restrict__ vna,
                   float* __restrict__ spc) {
  const int t = threadIdx.x;
  if (blockIdx.x < 96) spc[blockIdx.x * 128 + t] = 0.0f;   // 3*4096 floats

  const int row = blockIdx.x;
  const int mat = row >> 11;
  const int r   = row & (BS - 1);
  const float* src = (mat == 0) ? (z0  + (size_t)r * DIM)
                   : (mat == 1) ? (z1  + (size_t)r * DIM)
                                : (z1a + (size_t)r * DIM);
  float4 v = ((const float4*)src)[t];
  float ss = v.x * v.x + v.y * v.y + v.z * v.z + v.w * v.w;
  #pragma unroll
  for (int o = 16; o > 0; o >>= 1) ss += __shfl_xor(ss, o, 32);

  __shared__ float red[4];
  if ((t & 31) == 0) red[t >> 5] = ss;
  __syncthreads();
  const float scale = 1.0f / fmaxf(sqrtf(red[0] + red[1] + red[2] + red[3]), 1e-8f);

  v4h h;
  h.x = (_Float16)(v.x * scale);
  h.y = (_Float16)(v.y * scale);
  h.z = (_Float16)(v.z * scale);
  h.w = (_Float16)(v.w * scale);

  if (mat == 0) {
    ((v4h*)(vn  + (size_t)r * DIM))[t] = h;
    ((v4h*)(vna + (size_t)r * DIM))[t] = h;
  } else if (mat == 1) {
    ((v4h*)(vn  + (size_t)(BS + r) * DIM))[t] = h;
  } else {
    ((v4h*)(vna + (size_t)(BS + r) * DIM))[t] = h;
  }
}

// ---------------------------------------------------------------------------
// Phase 2: fused dual-GEMM + masked exp/sum partials.
// Grid (8 j-chunks, 32 row-panels) x 256 threads. Each wave owns 16 rows of a
// 128-row panel; all 8 waves share TDM-staged, double-buffered B tiles in LDS.
// ---------------------------------------------------------------------------
__global__ __launch_bounds__(256)
void sup_main(const _Float16* __restrict__ vn, const _Float16* __restrict__ vna,
              const int* __restrict__ labels,
              float* __restrict__ Sp, float* __restrict__ Pp,
              float* __restrict__ Cp) {
  __shared__ _Float16 lB[2][2][16][DIM];    // [buf][mat][col][k] = 64 KB

  const int tid  = threadIdx.x;
  const int wave = tid >> 5;
  const int lane = tid & 31;
  const int nloc = lane & 15;
  const int half = lane >> 4;
  const int jbase = blockIdx.x * 32;        // this WG's 32 column tiles
  const int rb    = blockIdx.y * 8 + wave;  // this wave's 16-row block

  // --- A fragments: 16 rows x 512 K x 2 matrices, resident in VGPRs -------
  v16h A0[16], A1[16];
  {
    const _Float16* ar0 = vn  + (size_t)(rb * 16 + nloc) * DIM + half * 16;
    const _Float16* ar1 = vna + (size_t)(rb * 16 + nloc) * DIM + half * 16;
    #pragma unroll
    for (int kb = 0; kb < 16; ++kb) {
      v8h l0 = *(const v8h*)(ar0 + kb * 32);
      v8h h0 = *(const v8h*)(ar0 + kb * 32 + 8);
      v8h l1 = *(const v8h*)(ar1 + kb * 32);
      v8h h1 = *(const v8h*)(ar1 + kb * 32 + 8);
      A0[kb] = __builtin_shufflevector(l0, h0, 0,1,2,3,4,5,6,7,8,9,10,11,12,13,14,15);
      A1[kb] = __builtin_shufflevector(l1, h1, 0,1,2,3,4,5,6,7,8,9,10,11,12,13,14,15);
    }
  }

  int labR[8];
  #pragma unroll
  for (int r = 0; r < 8; ++r)
    labR[r] = labels[(rb * 16 + half * 8 + r) & (BS - 1)];

  float Sacc[8], Pacc[8], Cacc[8];
  #pragma unroll
  for (int r = 0; r < 8; ++r) { Sacc[r] = 0.f; Pacc[r] = 0.f; Cacc[r] = 0.f; }

  // --- prologue: stage tile jbase into buffer 0 ---------------------------
#if USE_TDM
  if (wave == 0) {
    tdm_stage_16k(vn  + (size_t)jbase * 16 * DIM, &lB[0][0][0][0]);
    tdm_stage_16k(vna + (size_t)jbase * 16 * DIM, &lB[0][1][0][0]);
  }
#else
  stage_copy(vn  + (size_t)jbase * 16 * DIM,
             vna + (size_t)jbase * 16 * DIM, &lB[0][0][0][0], tid);
#endif

  for (int jj = 0; jj < 32; ++jj) {
    const int j   = jbase + jj;
    const int buf = jj & 1;

    // stage next tile into the other buffer, then fence current tile
#if USE_TDM
    if (wave == 0) {
      if (jj + 1 < 32) {
        tdm_stage_16k(vn  + (size_t)(j + 1) * 16 * DIM, &lB[buf ^ 1][0][0][0]);
        tdm_stage_16k(vna + (size_t)(j + 1) * 16 * DIM, &lB[buf ^ 1][1][0][0]);
        __builtin_amdgcn_s_wait_tensorcnt(2);   // current tile's 2 DMAs done
      } else {
        __builtin_amdgcn_s_wait_tensorcnt(0);
      }
    }
#else
    if (jj + 1 < 32)
      stage_copy(vn  + (size_t)(j + 1) * 16 * DIM,
                 vna + (size_t)(j + 1) * 16 * DIM, &lB[buf ^ 1][0][0][0], tid);
#endif
    __syncthreads();                            // publish current buffer

    // --- 16x16 tile: 2 matrices x K=512 -----------------------------------
    v8f acc0 = {0.f, 0.f, 0.f, 0.f, 0.f, 0.f, 0.f, 0.f};
    v8f acc1 = {0.f, 0.f, 0.f, 0.f, 0.f, 0.f, 0.f, 0.f};
    const _Float16* b0p = &lB[buf][0][nloc][half * 16];
    const _Float16* b1p = &lB[buf][1][nloc][half * 16];
    #pragma unroll
    for (int kb = 0; kb < 16; ++kb) {
      v16h b0 = *(const v16h*)(b0p + kb * 32);
      v16h b1 = *(const v16h*)(b1p + kb * 32);
      acc0 = __builtin_amdgcn_wmma_f32_16x16x32_f16(false, A0[kb], false, b0,
                                                    (short)0, acc0, false, false);
      acc1 = __builtin_amdgcn_wmma_f32_16x16x32_f16(false, A1[kb], false, b1,
                                                    (short)0, acc1, false, false);
    }

    const int gj   = j * 16 + nloc;
    const int labC = labels[gj & (BS - 1)];
    #pragma unroll
    for (int r = 0; r < 8; ++r) {               // C/D layout: VGPR r = row half*8+r
      const int gi = rb * 16 + half * 8 + r;
      const float s = (acc0[r] + acc1[r]) * (0.5f * INV_T);
      const bool offdiag = (gi != gj);
      Sacc[r] += offdiag ? __expf(s) : 0.0f;
      if (offdiag && (labR[r] == labC)) { Pacc[r] += s; Cacc[r] += 1.0f; }
    }
    __syncthreads();                            // release buffer for restage
  }

  // --- reduce over 16 columns per lane-half, then atomically merge --------
  #pragma unroll
  for (int r = 0; r < 8; ++r) {
    #pragma unroll
    for (int o = 1; o < 16; o <<= 1) {
      Sacc[r] += __shfl_xor(Sacc[r], o, 32);
      Pacc[r] += __shfl_xor(Pacc[r], o, 32);
      Cacc[r] += __shfl_xor(Cacc[r], o, 32);
    }
  }
  if (nloc == 0) {                              // lanes 0 and 16
    const int rowbase = rb * 16 + half * 8;
    #pragma unroll
    for (int r = 0; r < 8; ++r) {
      atomicAdd(&Sp[rowbase + r], Sacc[r]);
      atomicAdd(&Pp[rowbase + r], Pacc[r]);
      atomicAdd(&Cp[rowbase + r], Cacc[r]);
    }
  }
}

// ---------------------------------------------------------------------------
// Phase 3: loss = -mean_i( P_i/C_i - log S_i )
// ---------------------------------------------------------------------------
__global__ __launch_bounds__(256)
void sup_final(const float* __restrict__ Sp, const float* __restrict__ Pp,
               const float* __restrict__ Cp, float* __restrict__ out) {
  const int tid = threadIdx.x;
  float s = 0.f;
  for (int r = tid; r < NTOT; r += 256)
    s += Pp[r] / Cp[r] - __logf(Sp[r]);
  #pragma unroll
  for (int o = 16; o > 0; o >>= 1) s += __shfl_xor(s, o, 32);
  __shared__ float red[8];
  if ((tid & 31) == 0) red[tid >> 5] = s;
  __syncthreads();
  if (tid == 0) {
    float t = 0.f;
    #pragma unroll
    for (int w = 0; w < 8; ++w) t += red[w];
    out[0] = -t / (float)NTOT;
  }
}

// ---------------------------------------------------------------------------
extern "C" void kernel_launch(void* const* d_in, const int* in_sizes, int n_in,
                              void* d_out, int out_size, void* d_ws, size_t ws_size,
                              hipStream_t stream) {
  const float* z0  = (const float*)d_in[0];
  const float* z1  = (const float*)d_in[1];
  const float* z1a = (const float*)d_in[2];
  const int*   lab = (const int*)d_in[3];
  float* out = (float*)d_out;

  _Float16* vn  = (_Float16*)d_ws;                  // [4096][512] fp16
  _Float16* vna = vn + (size_t)NTOT * DIM;          // [4096][512] fp16
  float* spc = (float*)(vna + (size_t)NTOT * DIM);  // 3 x 4096 partials
  float* Sp = spc;
  float* Pp = spc + NTOT;
  float* Cp = spc + 2 * NTOT;

  sup_normalize<<<3 * BS, 128, 0, stream>>>(z0, z1, z1a, vn, vna, spc);
  dim3 grid(8, 32);
  sup_main<<<grid, 256, 0, stream>>>(vn, vna, lab, Sp, Pp, Cp);
  sup_final<<<1, 256, 0, stream>>>(Sp, Pp, Cp, out);
}
